// CausalAttention_60120952209614
// MI455X (gfx1250) — compile-verified, matrix operand order checked by reading, not run on hardware
//
#include <hip/hip_runtime.h>

typedef __attribute__((ext_vector_type(16))) _Float16 v16h;
typedef __attribute__((ext_vector_type(8)))  _Float16 v8h;
typedef __attribute__((ext_vector_type(8)))  float    v8f;
typedef int v4i __attribute__((vector_size(16)));   // matches async-builtin param type

#define T_CTX   2048
#define D_MODEL 2048
#define N_HEADS 16
#define HEAD_DIM 128
#define MTOT    (2 * T_CTX)          // 4096 rows (b*t)
#define N_QKV   (3 * D_MODEL)        // 6144
#define ATT_SCALE 0.08838834764831845f  // 1/sqrt(128)

#if __has_builtin(__builtin_amdgcn_global_load_async_to_lds_b128) && \
    __has_builtin(__builtin_amdgcn_s_wait_asynccnt)
#define HAVE_ASYNC_LDS 1
#else
#define HAVE_ASYNC_LDS 0
#endif

// 16-byte global -> LDS copy: async DMA (ASYNCcnt) on gfx1250 if available.
__device__ __forceinline__ void g2lds16(const _Float16* g, _Float16* l) {
#if HAVE_ASYNC_LDS
  __builtin_amdgcn_global_load_async_to_lds_b128(
      (__attribute__((address_space(1))) v4i*)g,
      (__attribute__((address_space(3))) v4i*)l, 0, 0);
#else
  *(v8h*)l = *(const v8h*)g;
#endif
}
__device__ __forceinline__ void async_fence() {
#if HAVE_ASYNC_LDS
  __builtin_amdgcn_s_wait_asynccnt(0);
#endif
}

__device__ __forceinline__ v16h combine8(v8h lo, v8h hi) {
  v16h r;
#pragma unroll
  for (int i = 0; i < 8; ++i) { r[i] = lo[i]; r[i + 8] = hi[i]; }
  return r;
}

// ---------------------------------------------------------------- convert
__global__ void cvt_f32_f16(const float* __restrict__ in,
                            _Float16* __restrict__ out, int n) {
  int i = blockIdx.x * blockDim.x + threadIdx.x;
  if (i < n) out[i] = (_Float16)in[i];
}

// ---------------------------------------------------------------- GEMM
// C[M][N] = A[M][K] * B[N][K]^T   (A,B f16 row-major, f32 accumulate)
// Double-buffered LDS; async global->LDS staging when available.
template <bool OUT_F32>
__global__ __launch_bounds__(256) void gemm_nt(
    const _Float16* __restrict__ A, const _Float16* __restrict__ B,
    void* __restrict__ Cout, const float* __restrict__ bias,
    int M, int N, int K) {
  __shared__ _Float16 sA[2][128 * 40];   // 128 rows x 32 (+8 pad)
  __shared__ _Float16 sB[2][128 * 40];

  const int tid   = threadIdx.x;
  const int wave  = tid >> 5;
  const int lane  = tid & 31;
  const int l16   = lane & 15;
  const int lhalf = lane >> 4;        // 0/1
  const int m0 = blockIdx.y * 128;
  const int n0 = blockIdx.x * 128;
  const int wr = wave & 3;            // M subtiles [2*wr, 2*wr+1]
  const int wc = wave >> 2;           // N subtiles [4*wc .. 4*wc+3]

  const int arow = tid >> 1;          // 0..127
  const int aseg = (tid & 1) * 16;    // 0 or 16 (f16 elems)

  auto stage = [&](int buf, int k0) {
    const _Float16* ga = A + (size_t)(m0 + arow) * K + k0 + aseg;
    const _Float16* gb = B + (size_t)(n0 + arow) * K + k0 + aseg;
    g2lds16(ga,     &sA[buf][arow * 40 + aseg]);
    g2lds16(ga + 8, &sA[buf][arow * 40 + aseg + 8]);
    g2lds16(gb,     &sB[buf][arow * 40 + aseg]);
    g2lds16(gb + 8, &sB[buf][arow * 40 + aseg + 8]);
  };

  v8f acc[2][4] = {};
  const int nk = K >> 5;              // K/32 tiles

  stage(0, 0);
  async_fence();
  __syncthreads();

  for (int t = 0; t < nk; ++t) {
    const int cur = t & 1;
    if (t + 1 < nk) stage(cur ^ 1, (t + 1) * 32);   // DMA next tile during compute

    v16h afrag[2], bfrag[4];
#pragma unroll
    for (int i = 0; i < 2; ++i) {
      int row = (wr * 2 + i) * 16 + l16;
      v8h lo = *(const v8h*)&sA[cur][row * 40 + lhalf * 8];
      v8h hi = *(const v8h*)&sA[cur][row * 40 + 16 + lhalf * 8];
      afrag[i] = combine8(lo, hi);
    }
#pragma unroll
    for (int j = 0; j < 4; ++j) {
      int row = (wc * 4 + j) * 16 + l16;
      v8h lo = *(const v8h*)&sB[cur][row * 40 + lhalf * 8];
      v8h hi = *(const v8h*)&sB[cur][row * 40 + 16 + lhalf * 8];
      bfrag[j] = combine8(lo, hi);
    }
#pragma unroll
    for (int i = 0; i < 2; ++i)
#pragma unroll
      for (int j = 0; j < 4; ++j)
        acc[i][j] = __builtin_amdgcn_wmma_f32_16x16x32_f16(
            false, afrag[i], false, bfrag[j], (short)0, acc[i][j], false, false);

    async_fence();     // next tile landed
    __syncthreads();   // everyone done reading cur
  }

  // epilogue: lane holds col n = l16, rows m = lhalf*8 + r
#pragma unroll
  for (int i = 0; i < 2; ++i) {
    int mbase = m0 + (wr * 2 + i) * 16 + lhalf * 8;
#pragma unroll
    for (int j = 0; j < 4; ++j) {
      int n = n0 + (wc * 4 + j) * 16 + l16;
#pragma unroll
      for (int r = 0; r < 8; ++r) {
        size_t off = (size_t)(mbase + r) * N + n;
        float v = acc[i][j][r];
        if (OUT_F32) {
          ((float*)Cout)[off] = v + (bias ? bias[n] : 0.0f);
        } else {
          ((_Float16*)Cout)[off] = (_Float16)v;
        }
      }
    }
  }
}

// ---------------------------------------------------------------- RoPE + reformat
__global__ void rope_reformat(const _Float16* __restrict__ qkv,
                              const float* __restrict__ cosT,
                              const float* __restrict__ sinT,
                              _Float16* __restrict__ Qh,
                              _Float16* __restrict__ Kh,
                              _Float16* __restrict__ Vh) {
  int idx = blockIdx.x * blockDim.x + threadIdx.x;  // 2*2048*16*64 = 2^22
  int d2 = idx & 63;
  int h  = (idx >> 6) & 15;
  int t  = (idx >> 10) & 2047;
  int b  = idx >> 21;
  size_t row = (size_t)(b * T_CTX + t) * N_QKV;
  int cb = h * HEAD_DIM;
  float c1 = cosT[t * HEAD_DIM + d2],      s1 = sinT[t * HEAD_DIM + d2];
  float c2 = cosT[t * HEAD_DIM + d2 + 64], s2 = sinT[t * HEAD_DIM + d2 + 64];
  float q1 = (float)qkv[row + cb + d2],             q2 = (float)qkv[row + cb + d2 + 64];
  float k1 = (float)qkv[row + D_MODEL + cb + d2],   k2 = (float)qkv[row + D_MODEL + cb + d2 + 64];
  float v1 = (float)qkv[row + 2*D_MODEL + cb + d2], v2 = (float)qkv[row + 2*D_MODEL + cb + d2 + 64];
  size_t ob = ((size_t)((b * N_HEADS + h) * T_CTX) + t) * HEAD_DIM;
  Qh[ob + d2]      = (_Float16)(q1 * c1 - q2 * s1);
  Qh[ob + d2 + 64] = (_Float16)(q2 * c2 + q1 * s2);
  Kh[ob + d2]      = (_Float16)(k1 * c1 - k2 * s1);
  Kh[ob + d2 + 64] = (_Float16)(k2 * c2 + k1 * s2);
  Vh[ob + d2]      = (_Float16)v1;
  Vh[ob + d2 + 64] = (_Float16)v2;
}

// ---------------------------------------------------------------- flash attention
// grid.x = b*h (32), grid.y = query blocks of 128. 8 waves, 16 queries each.
__global__ __launch_bounds__(256) void causal_attn(
    const _Float16* __restrict__ Qh, const _Float16* __restrict__ Kh,
    const _Float16* __restrict__ Vh, _Float16* __restrict__ ctxh) {
  __shared__ _Float16 sK[2][32 * 136];    // [key][d], pad to 136
  __shared__ _Float16 sV[2][128 * 40];    // [d][key], pad to 40

  const int tid   = threadIdx.x;
  const int wave  = tid >> 5;
  const int lane  = tid & 31;
  const int l16   = lane & 15;
  const int lhalf = lane >> 4;
  const int bh = blockIdx.x;           // b*16 + h
  const int b  = bh >> 4;
  const int h  = bh & 15;
  const int qblk = blockIdx.y * 128;
  const int qw   = qblk + wave * 16;   // wave's first query

  const _Float16* Qb = Qh + (size_t)bh * T_CTX * HEAD_DIM;
  const _Float16* Kb = Kh + (size_t)bh * T_CTX * HEAD_DIM;
  const _Float16* Vb = Vh + (size_t)bh * T_CTX * HEAD_DIM;

  // Q B-fragments (Q^T): lane = query column l16, 4 frags over head_dim
  v16h qfrag[4];
#pragma unroll
  for (int f = 0; f < 4; ++f) {
    const _Float16* qr = Qb + (size_t)(qw + l16) * HEAD_DIM + f * 32;
    v8h lo = *(const v8h*)(qr + lhalf * 8);
    v8h hi = *(const v8h*)(qr + 16 + lhalf * 8);
    qfrag[f] = combine8(lo, hi);
  }

  const int srow = tid >> 3;           // 0..31 staging key row
  const int sseg = (tid & 7) * 16;     // 0..112 d-offset

  auto stage = [&](int buf, int kb) {
    const _Float16* gk = Kb + (size_t)(kb + srow) * HEAD_DIM + sseg;
    g2lds16(gk,     &sK[buf][srow * 136 + sseg]);       // async K tile
    g2lds16(gk + 8, &sK[buf][srow * 136 + sseg + 8]);
    const v8h* gv = (const v8h*)(Vb + (size_t)(kb + srow) * HEAD_DIM + sseg);
    v8h v0 = gv[0], v1 = gv[1];                          // V transposed via VGPRs
#pragma unroll
    for (int i = 0; i < 8; ++i) {
      sV[buf][(sseg + i) * 40 + srow]     = v0[i];
      sV[buf][(sseg + 8 + i) * 40 + srow] = v1[i];
    }
  };

  v8f oacc[8] = {};                    // [head-dim tile][row]
  float mrun = -1e30f, lrun = 0.0f;
  const int niter = (qblk + 128) >> 5; // causal bound for whole block, /32

  stage(0, 0);
  async_fence();
  __syncthreads();

  for (int it = 0; it < niter; ++it) {
    const int kb  = it * 32;
    const int cur = it & 1;
    if (it + 1 < niter) stage(cur ^ 1, kb + 32);   // overlap fetch with compute

    // ---- preload all K A-fragments, then WMMA burst:
    // st[j] = K[kb+j*16..][.] x Q^T  (D: lane col = query l16)
    v16h kfrag[2][4];
#pragma unroll
    for (int j = 0; j < 2; ++j)
#pragma unroll
      for (int f = 0; f < 4; ++f) {
        int row = j * 16 + l16;
        v8h lo = *(const v8h*)&sK[cur][row * 136 + f * 32 + lhalf * 8];
        v8h hi = *(const v8h*)&sK[cur][row * 136 + f * 32 + 16 + lhalf * 8];
        kfrag[j][f] = combine8(lo, hi);
      }
    v8f st[2] = {};
#pragma unroll
    for (int j = 0; j < 2; ++j)
#pragma unroll
      for (int f = 0; f < 4; ++f)
        st[j] = __builtin_amdgcn_wmma_f32_16x16x32_f16(
            false, kfrag[j][f], false, qfrag[f], (short)0, st[j], false, false);

    // ---- scale + causal mask; online softmax (lane owns query q = qw + l16)
    const int qg = qw + l16;
    float s[2][8];
    float mloc = -1e30f;
#pragma unroll
    for (int j = 0; j < 2; ++j)
#pragma unroll
      for (int r = 0; r < 8; ++r) {
        int kg = kb + j * 16 + lhalf * 8 + r;
        float v = st[j][r] * ATT_SCALE;
        if (kg > qg) v = -1e30f;
        s[j][r] = v;
        mloc = fmaxf(mloc, v);
      }
    mloc = fmaxf(mloc, __shfl_xor(mloc, 16, 32));
    float mnew  = fmaxf(mrun, mloc);
    float alpha = __expf(mrun - mnew);

    float p[2][8];
    float lsum = 0.0f;
#pragma unroll
    for (int j = 0; j < 2; ++j)
#pragma unroll
      for (int r = 0; r < 8; ++r) {
        float e = __expf(s[j][r] - mnew);
        p[j][r] = e;
        lsum += e;
      }
    lsum += __shfl_xor(lsum, 16, 32);
    lrun = lrun * alpha + lsum;
    mrun = mnew;

    // rescale O: row r belongs to query (lhalf*8 + r) -> alpha at lane lhalf*8+r
    float arow[8];
#pragma unroll
    for (int r = 0; r < 8; ++r) arow[r] = __shfl(alpha, lhalf * 8 + r, 32);
#pragma unroll
    for (int ot = 0; ot < 8; ++ot)
#pragma unroll
      for (int r = 0; r < 8; ++r) oacc[ot][r] *= arow[r];

    // P as A-fragment: element e -> key (e<8?0:16) + lhalf*8 + (e&7) == p[e>=8][e&7]
    v16h pfrag;
#pragma unroll
    for (int e = 0; e < 16; ++e)
      pfrag[e] = (_Float16)(e < 8 ? p[0][e] : p[1][e - 8]);

    // ---- preload all V B-fragments, then WMMA burst: O += P x V
    v16h vfrag[8];
#pragma unroll
    for (int ot = 0; ot < 8; ++ot) {
      int drow = ot * 16 + l16;
      v8h lo = *(const v8h*)&sV[cur][drow * 40 + lhalf * 8];
      v8h hi = *(const v8h*)&sV[cur][drow * 40 + 16 + lhalf * 8];
      vfrag[ot] = combine8(lo, hi);
    }
#pragma unroll
    for (int ot = 0; ot < 8; ++ot)
      oacc[ot] = __builtin_amdgcn_wmma_f32_16x16x32_f16(
          false, pfrag, false, vfrag[ot], (short)0, oacc[ot], false, false);

    async_fence();
    __syncthreads();
  }

  // ---- normalize and write ctx [b][t][h*128+d] f16
  float linv = 1.0f / lrun;
  float lr[8];
#pragma unroll
  for (int r = 0; r < 8; ++r) lr[r] = __shfl(linv, lhalf * 8 + r, 32);
#pragma unroll
  for (int ot = 0; ot < 8; ++ot) {
    int d = h * HEAD_DIM + ot * 16 + l16;
#pragma unroll
    for (int r = 0; r < 8; ++r) {
      int qg = qw + lhalf * 8 + r;
      ctxh[((size_t)(b * T_CTX + qg)) * D_MODEL + d] = (_Float16)(oacc[ot][r] * lr[r]);
    }
  }
}

// ---------------------------------------------------------------- launch
extern "C" void kernel_launch(void* const* d_in, const int* in_sizes, int n_in,
                              void* d_out, int out_size, void* d_ws, size_t ws_size,
                              hipStream_t stream) {
  const float* x    = (const float*)d_in[0];
  const float* Wqkv = (const float*)d_in[1];
  const float* Wo   = (const float*)d_in[2];
  const float* bo   = (const float*)d_in[3];
  const float* cosT = (const float*)d_in[4];
  const float* sinT = (const float*)d_in[5];

  char* w = (char*)d_ws;
  _Float16* xh    = (_Float16*)(w);                    // 16 MB
  _Float16* wqkvh = (_Float16*)(w + 16777216);         // 24 MB
  _Float16* woh   = (_Float16*)(w + 41943040);         // 8 MB
  _Float16* qkvh  = (_Float16*)(w + 50331648);         // 48 MB
  _Float16* Qh    = (_Float16*)(w + 100663296);        // 16 MB
  _Float16* Kh    = (_Float16*)(w + 117440512);        // 16 MB
  _Float16* Vh    = (_Float16*)(w + 134217728);        // 16 MB
  _Float16* ctxh  = (_Float16*)(w + 150994944);        // 16 MB

  const int nx  = MTOT * D_MODEL;          // 8388608
  const int nwq = N_QKV * D_MODEL;         // 12582912
  const int nwo = D_MODEL * D_MODEL;       // 4194304
  cvt_f32_f16<<<(nx  + 255) / 256, 256, 0, stream>>>(x,    xh,    nx);
  cvt_f32_f16<<<(nwq + 255) / 256, 256, 0, stream>>>(Wqkv, wqkvh, nwq);
  cvt_f32_f16<<<(nwo + 255) / 256, 256, 0, stream>>>(Wo,   woh,   nwo);

  // qkv = x * Wqkv^T  -> f16 [4096][6144]
  gemm_nt<false><<<dim3(N_QKV / 128, MTOT / 128), 256, 0, stream>>>(
      xh, wqkvh, (void*)qkvh, nullptr, MTOT, N_QKV, D_MODEL);

  rope_reformat<<<(2 * T_CTX * N_HEADS * 64) / 256, 256, 0, stream>>>(
      qkvh, cosT, sinT, Qh, Kh, Vh);

  causal_attn<<<dim3(2 * N_HEADS, T_CTX / 128), 256, 0, stream>>>(Qh, Kh, Vh, ctxh);

  // out = ctx * Wo^T + b_o -> f32
  gemm_nt<true><<<dim3(D_MODEL / 128, MTOT / 128), 256, 0, stream>>>(
      ctxh, woh, d_out, bo, MTOT, D_MODEL, D_MODEL);
}
